// GCI_66211215835484
// MI455X (gfx1250) — matrix operand order repaired and checked by reading, not compiled, not run on hardware
//
#include <hip/hip_runtime.h>
#include <math.h>

#define NN 4096
#define BM 128
#define BN 128
#define BK 32
#define LDT 34   // ushort stride for LDS tiles (pad to dodge bank conflicts)
#define ATILE (BM * LDT)
#define BTILE (BN * LDT)

typedef __bf16  v16bf __attribute__((ext_vector_type(16)));
typedef float   v8f   __attribute__((ext_vector_type(8)));

// ---------- helpers ----------
// pack two f32 -> two bf16 (truncate) with a single v_perm_b32
__device__ __forceinline__ unsigned int pack2(float x, float y) {
    return __builtin_amdgcn_perm(__float_as_uint(y), __float_as_uint(x), 0x07060302u);
}
__device__ __forceinline__ unsigned short f2bf(float f) {
    return (unsigned short)(__float_as_uint(f) >> 16);   // truncate
}
// order-preserving float<->uint key (for atomic min/max over signed floats)
__device__ __forceinline__ unsigned int fkey(float f) {
    unsigned int b = __float_as_uint(f);
    return (b & 0x80000000u) ? ~b : (b | 0x80000000u);
}
__device__ __forceinline__ float funkey(unsigned int k) {
    unsigned int b = (k & 0x80000000u) ? (k & 0x7FFFFFFFu) : ~k;
    return __uint_as_float(b);
}

// ---------- software-pipelined staging: global -> regs, then regs -> LDS ----------
struct StageRegs { float4 a[4]; float4 b[4]; };

__device__ __forceinline__ void load_tiles(
    const float* __restrict__ A, const float* __restrict__ B, StageRegs& R,
    int M, int N, int K, int bT, int m0, int n0, int k0, int tid)
{
    {   // A tile: BMxBK
        int r  = tid >> 3;            // 0..31
        int cg = (tid & 7) * 4;       // 0..28
        #pragma unroll
        for (int p = 0; p < 4; ++p) {
            int gr = m0 + p * 32 + r;
            float4 v = {0.f,0.f,0.f,0.f};
            if (gr < M) v = *(const float4*)(A + (size_t)gr * K + k0 + cg);
            R.a[p] = v;
        }
    }
    if (bT) {                          // B is NxK: same access pattern as A
        int r  = tid >> 3;
        int cg = (tid & 7) * 4;
        #pragma unroll
        for (int p = 0; p < 4; ++p) {
            int gn = n0 + p * 32 + r;
            float4 v = {0.f,0.f,0.f,0.f};
            if (gn < N) v = *(const float4*)(B + (size_t)gn * K + k0 + cg);
            R.b[p] = v;
        }
    } else {                           // B is KxN: load 32xBN, transpose on LDS store
        int kr = tid >> 5;             // 0..7
        int cg = (tid & 31) * 4;       // 0..124
        #pragma unroll
        for (int p = 0; p < 4; ++p) {
            int gk = k0 + p * 8 + kr;
            int gn = n0 + cg;
            float4 v = {0.f,0.f,0.f,0.f};
            if (gn + 3 < N) {
                v = *(const float4*)(B + (size_t)gk * N + gn);
            } else {
                if (gn + 0 < N) v.x = B[(size_t)gk * N + gn + 0];
                if (gn + 1 < N) v.y = B[(size_t)gk * N + gn + 1];
                if (gn + 2 < N) v.z = B[(size_t)gk * N + gn + 2];
                if (gn + 3 < N) v.w = B[(size_t)gk * N + gn + 3];
            }
            R.b[p] = v;
        }
    }
}

__device__ __forceinline__ void store_tiles(
    const StageRegs& R, unsigned short* aS, unsigned short* bS, int bT, int tid)
{
    {   // A tile rows, K-contig
        int r  = tid >> 3;
        int cg = (tid & 7) * 4;
        #pragma unroll
        for (int p = 0; p < 4; ++p) {
            unsigned int* dst = (unsigned int*)&aS[(p * 32 + r) * LDT + cg];
            dst[0] = pack2(R.a[p].x, R.a[p].y);
            dst[1] = pack2(R.a[p].z, R.a[p].w);
        }
    }
    if (bT) {
        int r  = tid >> 3;
        int cg = (tid & 7) * 4;
        #pragma unroll
        for (int p = 0; p < 4; ++p) {
            unsigned int* dst = (unsigned int*)&bS[(p * 32 + r) * LDT + cg];
            dst[0] = pack2(R.b[p].x, R.b[p].y);
            dst[1] = pack2(R.b[p].z, R.b[p].w);
        }
    } else {
        int kr = tid >> 5;
        int cg = (tid & 31) * 4;
        #pragma unroll
        for (int p = 0; p < 4; ++p) {
            int kk = p * 8 + kr;
            bS[(cg + 0) * LDT + kk] = f2bf(R.b[p].x);
            bS[(cg + 1) * LDT + kk] = f2bf(R.b[p].y);
            bS[(cg + 2) * LDT + kk] = f2bf(R.b[p].z);
            bS[(cg + 3) * LDT + kk] = f2bf(R.b[p].w);
        }
    }
}

// ---------- bf16-WMMA tiled GEMM: C = [relu]( A(MxK) * B + bias ) ----------
// bT==0: B is KxN row-major.  bT==1: B is NxK row-major (pre-transposed).
// Double-buffered LDS + global->reg software pipeline; one barrier per K-step.
__global__ __launch_bounds__(256) void gemm_wmma_bf16(
    const float* __restrict__ A, const float* __restrict__ B,
    float* __restrict__ C, int M, int N, int K,
    int bT, const float* __restrict__ bias, int relu)
{
    __shared__ unsigned short aS[2][ATILE];
    __shared__ unsigned short bS[2][BTILE];

    const int tid  = threadIdx.x;
    const int lane = tid & 31;
    const int w    = tid >> 5;          // 8 waves
    const int wm   = w >> 1;            // 0..3  -> 32-row strip
    const int wn   = w & 1;             // 0..1  -> 64-col strip
    const int m0   = blockIdx.y * BM;
    const int n0   = blockIdx.x * BN;

    v8f zero = {0.f,0.f,0.f,0.f,0.f,0.f,0.f,0.f};
    v8f acc[2][4];
    #pragma unroll
    for (int i = 0; i < 2; ++i)
        #pragma unroll
        for (int j = 0; j < 4; ++j) acc[i][j] = zero;

    const int nk = K / BK;
    StageRegs R;
    load_tiles(A, B, R, M, N, K, bT, m0, n0, 0, tid);
    store_tiles(R, aS[0], bS[0], bT, tid);
    __syncthreads();

    for (int kt = 0; kt < nk; ++kt) {
        const int cur = kt & 1;
        // issue global loads for the next tile (no wait yet)
        if (kt + 1 < nk)
            load_tiles(A, B, R, M, N, K, bT, m0, n0, (kt + 1) * BK, tid);
        // warm L2 for the tile after next (global_prefetch_b8)
        if (kt + 2 < nk) {
            int k2 = (kt + 2) * BK;
            int gr = m0 + (tid >> 3);
            if (gr < M) __builtin_prefetch(A + (size_t)gr * K + k2, 0, 3);
        }

        // ---- build fragments per CDNA5 wave32 WMMA layouts ----
        union FragU { v16bf v; unsigned int u[8]; } fa[2], fb[4];
        {
            const unsigned short* aC = aS[cur];
            const unsigned short* bC = bS[cur];
            int m  = lane & 15;
            int kb = (lane >> 4) * 8;     // lanes 16-31 -> K base +8
            #pragma unroll
            for (int mt = 0; mt < 2; ++mt) {
                const unsigned short* rp = &aC[(wm * 32 + mt * 16 + m) * LDT];
                #pragma unroll
                for (int v = 0; v < 4; ++v)
                    fa[mt].u[v] = *(const unsigned int*)(rp + kb + 2 * v);
                #pragma unroll
                for (int v = 0; v < 4; ++v)
                    fa[mt].u[4 + v] = *(const unsigned int*)(rp + 16 + kb + 2 * v);
            }
            int nn2 = lane & 15;
            int kh  = (lane >> 4) * 8;    // lanes 16-31 -> K base +8
            #pragma unroll
            for (int nt = 0; nt < 4; ++nt) {
                const unsigned short* cp = &bC[(wn * 64 + nt * 16 + nn2) * LDT + kh];
                #pragma unroll
                for (int v = 0; v < 8; ++v)
                    fb[nt].u[v] = *(const unsigned int*)(cp + 2 * v + (v >> 2) * 8);
            }
        }
        #pragma unroll
        for (int mt = 0; mt < 2; ++mt)
            #pragma unroll
            for (int nt = 0; nt < 4; ++nt)
                acc[mt][nt] = __builtin_amdgcn_wmma_f32_16x16x32_bf16(
                    false, fa[mt].v, false, fb[nt].v,
                    (short)0, acc[mt][nt], false, false);

        // convert + store next tile (loadcnt wait lands here, after the WMMAs)
        if (kt + 1 < nk)
            store_tiles(R, aS[cur ^ 1], bS[cur ^ 1], bT, tid);
        __syncthreads();
    }

    // ---- epilogue: C/D layout M = vgpr + (lane>=16)*8, N = lane&15 ----
    {
        int rb = (lane >> 4) * 8;
        int cn = lane & 15;
        #pragma unroll
        for (int mt = 0; mt < 2; ++mt) {
            int rowb = m0 + wm * 32 + mt * 16 + rb;
            #pragma unroll
            for (int nt = 0; nt < 4; ++nt) {
                int col = n0 + wn * 64 + nt * 16 + cn;
                if (col >= N) continue;
                float bv = bias ? bias[col] : 0.f;
                #pragma unroll
                for (int v = 0; v < 8; ++v) {
                    int row = rowb + v;
                    if (row >= M) continue;
                    float x = acc[mt][nt][v] + bv;
                    if (relu && x < 0.f) x = 0.f;
                    C[(size_t)row * N + col] = x;
                }
            }
        }
    }
}

// ---------- init: zero stats + histograms, seed min/max keys ----------
__global__ void k_init(unsigned int* wsu) {
    int total = 64 + 65536 * 2;
    for (int i = blockIdx.x * blockDim.x + threadIdx.x; i < total;
         i += gridDim.x * blockDim.x) wsu[i] = 0u;
    if (blockIdx.x == 0 && threadIdx.x == 0) wsu[0] = 0xFFFFFFFFu;  // min key
}

// ---------- Z = noise * exp(logstd) + mean ----------
__global__ void k_z(const float* __restrict__ noise, const float* __restrict__ logstd,
                    const float* __restrict__ mean, float* __restrict__ Z, int n) {
    for (int i = blockIdx.x * blockDim.x + threadIdx.x; i < n;
         i += gridDim.x * blockDim.x)
        Z[i] = noise[i] * __expf(logstd[i]) + mean[i];
}

// ---------- min/max of strict-upper logits + count(adj!=0) ----------
__global__ __launch_bounds__(256) void k_minmax_count(
    const float* __restrict__ logits, const float* __restrict__ adj,
    unsigned int* stats)
{
    __shared__ unsigned int sMin[256], sMax[256], sCnt[256];
    unsigned int mn = 0xFFFFFFFFu, mx = 0u, cnt = 0;
    size_t total = (size_t)NN * NN;
    for (size_t idx = (size_t)blockIdx.x * blockDim.x + threadIdx.x; idx < total;
         idx += (size_t)gridDim.x * blockDim.x) {
        int i = (int)(idx >> 12), j = (int)(idx & (NN - 1));
        float L = logits[idx];
        if (j > i) {
            unsigned int k = fkey(L);
            mn = (k < mn) ? k : mn;
            mx = (k > mx) ? k : mx;
        }
        if (adj[idx] != 0.f) cnt++;
    }
    sMin[threadIdx.x] = mn; sMax[threadIdx.x] = mx; sCnt[threadIdx.x] = cnt;
    __syncthreads();
    for (int s = 128; s > 0; s >>= 1) {
        if (threadIdx.x < (unsigned)s) {
            unsigned int a = sMin[threadIdx.x + s];
            if (a < sMin[threadIdx.x]) sMin[threadIdx.x] = a;
            unsigned int b = sMax[threadIdx.x + s];
            if (b > sMax[threadIdx.x]) sMax[threadIdx.x] = b;
            sCnt[threadIdx.x] += sCnt[threadIdx.x + s];
        }
        __syncthreads();
    }
    if (threadIdx.x == 0) {
        atomicMin(&stats[0], sMin[0]);
        atomicMax(&stats[1], sMax[0]);
        atomicAdd(&stats[2], sCnt[0]);
    }
}

// ---------- derive gmin/gmax of ep (triu includes zeros) ----------
__global__ void k_range(unsigned int* su, float* sf) {
    float mnU = funkey(su[0]);
    float mxU = funkey(su[1]);
    float gmin = fminf(mnU, 0.f);
    float gmax = fmaxf(mxU, 0.f) - gmin;
    if (!(gmax > 0.f)) gmax = 1.f;
    sf[7] = gmin; sf[8] = gmax;
}

// ---------- histograms of mask_rm / mask_add positives ----------
__global__ __launch_bounds__(256) void k_hist(
    const float* __restrict__ logits, const float* __restrict__ adj,
    const float* __restrict__ sf, unsigned int* __restrict__ hist_rm,
    unsigned int* __restrict__ hist_add)
{
    float gmin = sf[7], ginv = 1.f / sf[8];
    size_t total = (size_t)NN * NN;
    for (size_t idx = (size_t)blockIdx.x * blockDim.x + threadIdx.x; idx < total;
         idx += (size_t)gridDim.x * blockDim.x) {
        int i = (int)(idx >> 12), j = (int)(idx & (NN - 1));
        float L  = logits[idx];
        float ep = (((j > i) ? L : 0.f) - gmin) * ginv;     // ep in [0,1]
        float a  = adj[idx];
        float mrm = ep * a;
        if (mrm > 0.f) {
            int b = (int)(mrm * 65536.f); if (b > 65535) b = 65535;
            atomicAdd(&hist_rm[b], 1u);
        }
        float mad = ep * (1.f - a);
        if (mad > 0.f) {
            int b = (int)(mad * 65536.f); if (b > 65535) b = 65535;
            atomicAdd(&hist_add[b], 1u);
        }
    }
}

// ---------- scan histograms -> thresholds ----------
__global__ void k_thresh(const unsigned int* su, float* sf,
                         const unsigned int* hist_rm, const unsigned int* hist_add)
{
    unsigned long long nnz_rm = 0, nnz_add = 0;
    for (int b = 0; b < 65536; ++b) { nnz_rm += hist_rm[b]; nnz_add += hist_add[b]; }
    unsigned long long n_change = (unsigned long long)(su[2] >> 1);
    unsigned long long n_rm  = (nnz_rm  < n_change) ? nnz_rm  : n_change;
    unsigned long long n_add = (nnz_add < n_change) ? nnz_add : n_change;
    float thr_rm = 0.f;        // keep 0 < v <= thr_rm
    if (n_rm > 0) {
        unsigned long long c = 0;
        for (int b = 0; b < 65536; ++b) {
            c += hist_rm[b];
            if (c >= n_rm) { thr_rm = (float)(b + 1) * (1.f / 65536.f); break; }
        }
    }
    float thr_add = 2.f;       // keep v >= thr_add (2.0 => keep none)
    if (n_add > 0) {
        unsigned long long c = 0;
        for (int b = 65535; b >= 0; --b) {
            c += hist_add[b];
            if (c >= n_add) { thr_add = (float)b * (1.f / 65536.f); break; }
        }
    }
    sf[5] = thr_rm; sf[6] = thr_add;
}

// ---------- fused adj_new build + L1 row normalize + row sums ----------
__global__ __launch_bounds__(256) void k_adjnew(
    const float* __restrict__ logits, const float* __restrict__ adj,
    const float* __restrict__ sf, float* __restrict__ adj_new,
    float* __restrict__ gsum)
{
    __shared__ float sAbs[256], sSig[256];
    int i = blockIdx.x;
    float gmin = sf[7], ginv = 1.f / sf[8], thr_rm = sf[5], thr_add = sf[6];
    float accA = 0.f, accS = 0.f;
    for (int j = threadIdx.x; j < NN; j += 256) {
        size_t idx = (size_t)i * NN + j;
        float L = logits[idx];                  // symmetric: L(i,j)==L(j,i)
        float a = adj[idx];
        float v;
        if (j == i) v = 1.f;
        else {
            float ep_ij = (((j > i) ? L : 0.f) - gmin) * ginv;
            float ep_ji = (((i > j) ? L : 0.f) - gmin) * ginv;
            float mrm_ij = ep_ij * a,          mrm_ji = ep_ji * a;
            float mad_ij = ep_ij * (1.f - a),  mad_ji = ep_ji * (1.f - a);
            float rm = ((mrm_ij > 0.f && mrm_ij <= thr_rm) ? 1.f : 0.f)
                     + ((mrm_ji > 0.f && mrm_ji <= thr_rm) ? 1.f : 0.f);
            float ad = ((mad_ij > 0.f && mad_ij >= thr_add) ? 1.f : 0.f)
                     + ((mad_ji > 0.f && mad_ji >= thr_add) ? 1.f : 0.f);
            v = a - rm + ad;
        }
        adj_new[idx] = v;
        accA += fabsf(v); accS += v;
    }
    sAbs[threadIdx.x] = accA; sSig[threadIdx.x] = accS;
    __syncthreads();
    for (int s = 128; s > 0; s >>= 1) {
        if (threadIdx.x < (unsigned)s) {
            sAbs[threadIdx.x] += sAbs[threadIdx.x + s];
            sSig[threadIdx.x] += sSig[threadIdx.x + s];
        }
        __syncthreads();
    }
    float inv = 1.f / fmaxf(sAbs[0], 1e-12f);
    for (int j = threadIdx.x; j < NN; j += 256)
        adj_new[(size_t)i * NN + j] *= inv;
    if (threadIdx.x == 0) gsum[i] = sSig[0] * inv;
}

// ---------- concat builders ----------
__global__ void k_concat1(const float* __restrict__ feats, const float* __restrict__ neigh,
                          const float* __restrict__ gsum, float* __restrict__ out) {
    int total = NN * 1024;
    for (int idx = blockIdx.x * blockDim.x + threadIdx.x; idx < total;
         idx += gridDim.x * blockDim.x) {
        int i = idx >> 10, c = idx & 1023;
        out[idx] = (c < 512) ? feats[(size_t)i * 512 + c]
                             : neigh[(size_t)i * 512 + (c - 512)] / (gsum[i] + 1e-7f);
    }
}
__global__ void k_concat2(const float* __restrict__ h1, const float* __restrict__ neigh2,
                          const float* __restrict__ gsum, float* __restrict__ out) {
    int total = NN * 512;
    for (int idx = blockIdx.x * blockDim.x + threadIdx.x; idx < total;
         idx += gridDim.x * blockDim.x) {
        int i = idx >> 9, c = idx & 511;
        out[idx] = (c < 256) ? h1[(size_t)i * 256 + c]
                             : neigh2[(size_t)i * 256 + (c - 256)] / (gsum[i] + 1e-7f);
    }
}

// ---------- in-place row log_softmax over 512 cols ----------
__global__ __launch_bounds__(256) void k_logsoftmax(float* __restrict__ x) {
    __shared__ float sred[256];
    float* row = x + (size_t)blockIdx.x * 512;
    float mx = -1e30f;
    for (int c = threadIdx.x; c < 512; c += 256) mx = fmaxf(mx, row[c]);
    sred[threadIdx.x] = mx; __syncthreads();
    for (int s = 128; s > 0; s >>= 1) {
        if (threadIdx.x < (unsigned)s)
            sred[threadIdx.x] = fmaxf(sred[threadIdx.x], sred[threadIdx.x + s]);
        __syncthreads();
    }
    mx = sred[0]; __syncthreads();
    float se = 0.f;
    for (int c = threadIdx.x; c < 512; c += 256) se += __expf(row[c] - mx);
    sred[threadIdx.x] = se; __syncthreads();
    for (int s = 128; s > 0; s >>= 1) {
        if (threadIdx.x < (unsigned)s)
            sred[threadIdx.x] += sred[threadIdx.x + s];
        __syncthreads();
    }
    float lse = mx + __logf(sred[0]);
    for (int c = threadIdx.x; c < 512; c += 256) row[c] -= lse;
}

// ================= host side =================
extern "C" void kernel_launch(void* const* d_in, const int* in_sizes, int n_in,
                              void* d_out, int out_size, void* d_ws, size_t ws_size,
                              hipStream_t stream) {
    const float* adj_norm = (const float*)d_in[0];
    const float* adj_ori  = (const float*)d_in[1];
    const float* feats    = (const float*)d_in[2];
    const float* img      = (const float*)d_in[3];
    const float* csd_img  = (const float*)d_in[5];
    const float* noise    = (const float*)d_in[6];
    const float* Wb   = (const float*)d_in[7];
    const float* Wm   = (const float*)d_in[8];
    const float* Wl   = (const float*)d_in[9];
    const float* fc1W = (const float*)d_in[10];
    const float* fc1b = (const float*)d_in[11];
    const float* fc2W = (const float*)d_in[12];
    const float* fc2b = (const float*)d_in[13];

    float* out = (float*)d_out;
    float* preds_total = out;                                   // 4096*64
    float* feat_total  = out + 262144;                          // 4096*512
    float* preds_img   = out + 262144 + 2097152;                // 4096*64
    float* adj_logits  = out + 262144 + 2097152 + 262144;       // 4096*4096

    float*        wsf = (float*)d_ws;
    unsigned int* wsu = (unsigned int*)d_ws;
    unsigned int* hist_rm  = wsu + 64;
    unsigned int* hist_add = wsu + 64 + 65536;
    float* gsum = wsf + 64 + 2 * 65536;                         // 4096
    size_t off = 64 + 2 * 65536 + 4096;
    float* T1     = wsf + off; off += (size_t)NN * 256;
    float* HID    = wsf + off; off += (size_t)NN * 256;
    float* T2     = wsf + off; off += (size_t)NN * 128;
    float* T3     = wsf + off; off += (size_t)NN * 128;
    float* MEAN   = wsf + off; off += (size_t)NN * 128;
    float* LOGSTD = wsf + off; off += (size_t)NN * 128;
    float* Zb     = wsf + off; off += (size_t)NN * 128;
    float* ADJ    = wsf + off; off += (size_t)NN * NN;
    float* NEIGH  = wsf + off; off += (size_t)NN * 512;
    float* CONCAT = wsf + off; off += (size_t)NN * 1024;
    float* H1     = wsf + off; off += (size_t)NN * 256;

    dim3 blk(256);
    auto gemm = [&](const float* A, const float* B, float* C, int M, int N, int K,
                    int bT, const float* bias, int relu) {
        dim3 grid((N + BN - 1) / BN, (M + BM - 1) / BM);
        gemm_wmma_bf16<<<grid, blk, 0, stream>>>(A, B, C, M, N, K, bT, bias, relu);
    };

    k_init<<<256, blk, 0, stream>>>(wsu);

    // ep_net (VGAE)
    gemm(img, Wb, T1, NN, 256, 512, 0, nullptr, 0);             // img @ Wb
    gemm(adj_norm, T1, HID, NN, 256, NN, 0, nullptr, 0);        // hidden
    gemm(HID, Wm, T2, NN, 128, 256, 0, nullptr, 0);
    gemm(adj_norm, T2, MEAN, NN, 128, NN, 0, nullptr, 1);       // relu
    gemm(HID, Wl, T3, NN, 128, 256, 0, nullptr, 0);
    gemm(adj_norm, T3, LOGSTD, NN, 128, NN, 0, nullptr, 1);     // relu
    k_z<<<2048, blk, 0, stream>>>(noise, LOGSTD, MEAN, Zb, NN * 128);
    gemm(Zb, Zb, adj_logits, NN, NN, 128, 1, nullptr, 0);       // Z @ Z^T

    // edge rewiring (histogram-based k-th selection instead of full sort)
    k_minmax_count<<<8192, blk, 0, stream>>>(adj_logits, adj_ori, wsu);
    k_range<<<1, 1, 0, stream>>>(wsu, wsf);
    k_hist<<<8192, blk, 0, stream>>>(adj_logits, adj_ori, wsf, hist_rm, hist_add);
    k_thresh<<<1, 1, 0, stream>>>(wsu, wsf, hist_rm, hist_add);
    k_adjnew<<<NN, blk, 0, stream>>>(adj_logits, adj_ori, wsf, ADJ, gsum);

    // nc_net (DGPN / GraphSAGE)
    gemm(ADJ, feats, NEIGH, NN, 512, NN, 0, nullptr, 0);        // adj_new @ feats
    k_concat1<<<4096, blk, 0, stream>>>(feats, NEIGH, gsum, CONCAT);
    gemm(CONCAT, fc1W, H1, NN, 256, 1024, 0, fc1b, 1);          // h1 (relu)
    gemm(ADJ, H1, NEIGH, NN, 256, NN, 0, nullptr, 0);           // adj_new @ h1
    k_concat2<<<4096, blk, 0, stream>>>(H1, NEIGH, gsum, CONCAT);
    gemm(CONCAT, fc2W, feat_total, NN, 512, 512, 0, fc2b, 0);   // h2
    k_logsoftmax<<<NN, blk, 0, stream>>>(feat_total);
    gemm(feat_total, csd_img, preds_total, NN, 64, 512, 1, nullptr, 0);
    gemm(img, csd_img, preds_img, NN, 64, 512, 1, nullptr, 0);
}
